// Encoder_75831942578735
// MI455X (gfx1250) — matrix operand order repaired and checked by reading, not compiled, not run on hardware
//
#include <hip/hip_runtime.h>

// ---------------------------------------------------------------------------
// GCN encoder for MI455X (gfx1250, wave32).
// GEMMs via v_wmma_f32_16x16x32_bf16 with split-bf16 (hi/lo) for fp32 accuracy.
// Weights are pre-swizzled into WMMA-fragment order so each lane loads its
// B fragment with two b128 global loads (L2-resident, no LDS, no packing).
// Edge scatter via hardware f32 global atomics; GraphNorm stats use
// block-local accumulation (batch is sorted) to minimize atomic contention.
// ---------------------------------------------------------------------------

typedef __attribute__((ext_vector_type(16))) __bf16          v16bf;
typedef __attribute__((ext_vector_type(16))) unsigned short  v16us;
typedef __attribute__((ext_vector_type(8)))  unsigned int    v8ui;
typedef __attribute__((ext_vector_type(8)))  float           v8f;

#define EPS 1e-5f

// ---------------- small utility kernels ----------------
__global__ void fill_f32(float* __restrict__ p, float v, int n) {
    int t = blockIdx.x * blockDim.x + threadIdx.x;
    if (t < n) p[t] = v;
}

// Split fp32 weight into bf16 hi/lo AND swizzle into WMMA B-fragment order:
// hi[((kstep*4 + nt)*32 + lane)*16 + e] = bf16(W[k][n]),
//   k = kstep*32 + (e<8 ? (lane>>4)*8 + e : 16 + (lane>>4)*8 + e-8),  n = nt*16 + (lane&15).
// Each lane's 16 elements land contiguously (32 bytes) -> 2x b128 loads.
__global__ void split_swizzle_weights(const float* __restrict__ W,
                                      unsigned short* __restrict__ hi,
                                      unsigned short* __restrict__ lo, int K) {
    int t = blockIdx.x * blockDim.x + threadIdx.x;
    if (t >= K * 64) return;
    int e     = t & 15;
    int lane  = (t >> 4) & 31;
    int nt    = (t >> 9) & 3;
    int kstep = t >> 11;
    int k = kstep * 32 + (e < 8 ? (lane >> 4) * 8 + e : 16 + (lane >> 4) * 8 + (e - 8));
    int n = nt * 16 + (lane & 15);
    float v = W[k * 64 + n];
    unsigned int u = __float_as_uint(v) & 0xFFFF0000u;
    hi[t] = (unsigned short)(u >> 16);
    lo[t] = (unsigned short)(__float_as_uint(v - __uint_as_float(u)) >> 16);
}

// deg[dst] += 1 per edge (deg pre-filled with 1.0 for the self-loop).
__global__ void edge_degree(const int* __restrict__ dst, float* __restrict__ deg, int E) {
    int t = blockIdx.x * blockDim.x + threadIdx.x;
    if (t < E) atomicAdd(&deg[dst[t]], 1.0f);
}

__global__ void finalize_dinv(float* __restrict__ d, int N) {
    int t = blockIdx.x * blockDim.x + threadIdx.x;
    if (t < N) { float v = d[t]; d[t] = (v > 0.0f) ? rsqrtf(v) : 0.0f; }
}

// Pack 8 floats (as two bf16-split streams) into packed-dword hi/lo halves.
__device__ __forceinline__ void pack4(const float4 a, const float4 b,
                                      unsigned int* hiw, unsigned int* low) {
    const float vs[8] = {a.x, a.y, a.z, a.w, b.x, b.y, b.z, b.w};
#pragma unroll
    for (int j = 0; j < 4; ++j) {
        unsigned int u0 = __float_as_uint(vs[2 * j]);
        unsigned int u1 = __float_as_uint(vs[2 * j + 1]);
        unsigned int h0 = u0 & 0xFFFF0000u;
        unsigned int h1 = u1 & 0xFFFF0000u;
        hiw[j] = (h0 >> 16) | h1;
        float l0 = vs[2 * j]     - __uint_as_float(h0);
        float l1 = vs[2 * j + 1] - __uint_as_float(h1);
        low[j] = (__float_as_uint(l0) >> 16) | (__float_as_uint(l1) & 0xFFFF0000u);
    }
}

// ---------------- WMMA GEMM: Y[M,64] = X[M,K] @ W[K,64] ----------------
// One wave -> 16x64 output tile (4 n-tiles of 16x16). 8 waves / 256-thread block.
// 3-product split-bf16: acc += Ahi*Bhi + Alo*Bhi + Ahi*Blo.
template <int K>
__global__ __launch_bounds__(256)
void gemm_wmma_bf16split(const float* __restrict__ X,
                         const unsigned short* __restrict__ Whi,
                         const unsigned short* __restrict__ Wlo,
                         float* __restrict__ Y, int M) {
    const int wave = threadIdx.x >> 5;            // 0..7
    const int lane = threadIdx.x & 31;
    const int rowTile = blockIdx.x * 8 + wave;    // 16 rows per tile
    const long row0 = (long)rowTile * 16;
    if (row0 >= M) return;                        // wave-uniform exit

    const int m    = lane & 15;
    const int half = lane >> 4;                   // 0 or 1
    long row = row0 + m;
    if (row >= M) row = M - 1;                    // clamp loads, stores guarded

    const v16us* __restrict__ bHiP = (const v16us*)Whi;
    const v16us* __restrict__ bLoP = (const v16us*)Wlo;

    v8f acc[4] = {v8f{}, v8f{}, v8f{}, v8f{}};

#pragma unroll
    for (int ks = 0; ks < K; ks += 32) {
        // ---- A fragment (ISA 16-bit A layout):
        // lane half 0: K = ks+{0..7, 16..23}; half 1: K = ks+{8..15, 24..31}
        const float* xr = X + row * K + ks + half * 8;
        float4 p0 = *(const float4*)(xr);
        float4 p1 = *(const float4*)(xr + 4);
        float4 p2 = *(const float4*)(xr + 16);
        float4 p3 = *(const float4*)(xr + 20);
        v8ui aHiU, aLoU;
        pack4(p0, p1, (unsigned int*)&aHiU,     (unsigned int*)&aLoU);
        pack4(p2, p3, (unsigned int*)&aHiU + 4, (unsigned int*)&aLoU + 4);
        v16bf aHi = __builtin_bit_cast(v16bf, aHiU);
        v16bf aLo = __builtin_bit_cast(v16bf, aLoU);

        const int kstep = ks >> 5;
#pragma unroll
        for (int nt = 0; nt < 4; ++nt) {
            // ---- B fragment: pre-swizzled, 32B contiguous per lane
            const int frag = (kstep * 4 + nt) * 32 + lane;
            v16bf bHi = __builtin_bit_cast(v16bf, bHiP[frag]);
            v16bf bLo = __builtin_bit_cast(v16bf, bLoP[frag]);
            acc[nt] = __builtin_amdgcn_wmma_f32_16x16x32_bf16(
                false, aHi, false, bHi, (short)0, acc[nt], false, false);
            acc[nt] = __builtin_amdgcn_wmma_f32_16x16x32_bf16(
                false, aLo, false, bHi, (short)0, acc[nt], false, false);
            acc[nt] = __builtin_amdgcn_wmma_f32_16x16x32_bf16(
                false, aHi, false, bLo, (short)0, acc[nt], false, false);
        }
    }

    // ---- C/D layout: lane n = lane&15, VGPR r holds row m = r + 8*(lane>>4)
    const int n = lane & 15;
#pragma unroll
    for (int r = 0; r < 8; ++r) {
        long mrow = row0 + r + half * 8;
        if (mrow < M) {
#pragma unroll
            for (int nt = 0; nt < 4; ++nt)
                Y[mrow * 64 + nt * 16 + n] = acc[nt][r];
        }
    }
}

// ---------------- GCN propagation ----------------
// out[i] = h[i]*dinv[i]^2 + bias  (self-loop term, initializes the buffer)
__global__ void scatter_init(const float* __restrict__ h, const float* __restrict__ dinv,
                             const float* __restrict__ bias, float* __restrict__ out, int N) {
    long t = (long)blockIdx.x * blockDim.x + threadIdx.x;
    long i = t >> 4;
    if (i >= N) return;
    int f = (int)(t & 15) * 4;
    float di = dinv[i];
    float s  = di * di;
    const float4 hv = *(const float4*)(h + i * 64 + f);
    float4 o;
    o.x = hv.x * s + bias[f + 0];
    o.y = hv.y * s + bias[f + 1];
    o.z = hv.z * s + bias[f + 2];
    o.w = hv.w * s + bias[f + 3];
    *(float4*)(out + i * 64 + f) = o;
}

// out[dst] += h[src] * dinv[src] * dinv[dst], 16 threads/edge, float4 each
__global__ void scatter_edges(const float* __restrict__ h, const int* __restrict__ src,
                              const int* __restrict__ dst, const float* __restrict__ dinv,
                              float* __restrict__ out, int E) {
    long t = (long)blockIdx.x * blockDim.x + threadIdx.x;
    long e = t >> 4;
    if (e >= E) return;
    int f = (int)(t & 15) * 4;
    int s = src[e], d = dst[e];
    float nrm = dinv[s] * dinv[d];
    const float4 hv = *(const float4*)(h + (long)s * 64 + f);
    float* o = out + (long)d * 64 + f;
    atomicAdd(o + 0, hv.x * nrm);
    atomicAdd(o + 1, hv.y * nrm);
    atomicAdd(o + 2, hv.z * nrm);
    atomicAdd(o + 3, hv.w * nrm);
}

// ---------------- GraphNorm ----------------
// batch is sorted: accumulate locally, flush atomics only at graph boundaries.
__global__ __launch_bounds__(256)
void graph_stats(const float* __restrict__ h, const int* __restrict__ batch,
                 float* __restrict__ s1, float* __restrict__ s2,
                 float* __restrict__ cnt, int N) {
    const int f   = threadIdx.x & 63;
    const int grp = threadIdx.x >> 6;   // 4 node-groups per block
    const long base = (long)blockIdx.x * 256;
    float a1 = 0.0f, a2 = 0.0f, c = 0.0f;
    int curg = -1;
    for (int j = grp; j < 256; j += 4) {
        long i = base + j;
        if (i >= N) break;
        int g = batch[i];
        if (g != curg) {
            if (curg >= 0) {
                atomicAdd(&s1[curg * 64 + f], a1);
                atomicAdd(&s2[curg * 64 + f], a2);
                if (f == 0) atomicAdd(&cnt[curg], c);
            }
            curg = g; a1 = 0.0f; a2 = 0.0f; c = 0.0f;
        }
        float v = h[i * 64 + f];
        a1 += v; a2 += v * v; c += 1.0f;
    }
    if (curg >= 0) {
        atomicAdd(&s1[curg * 64 + f], a1);
        atomicAdd(&s2[curg * 64 + f], a2);
        if (f == 0) atomicAdd(&cnt[curg], c);
    }
}

// out = relu(weight*(x - alpha*mean)*rsqrt(var+eps) + bias)
// var = E[x^2] - (2a - a^2) * mean^2   (since centered = x - a*mean)
__global__ void norm_relu(const float* __restrict__ h, const int* __restrict__ batch,
                          const float* __restrict__ s1, const float* __restrict__ s2,
                          const float* __restrict__ cnt, const float* __restrict__ alpha,
                          const float* __restrict__ weight, const float* __restrict__ bias,
                          float* __restrict__ out, int N) {
    long t = (long)blockIdx.x * blockDim.x + threadIdx.x;
    long i = t >> 6;
    if (i >= N) return;
    int f = (int)(t & 63);
    int g = batch[i];
    float n    = fmaxf(cnt[g], 1.0f);
    float mean = s1[g * 64 + f] / n;
    float ex2  = s2[g * 64 + f] / n;
    float a    = alpha[f];
    float var  = ex2 - (2.0f * a - a * a) * mean * mean;
    float inv  = rsqrtf(var + EPS);
    float v    = weight[f] * (h[i * 64 + f] - a * mean) * inv + bias[f];
    out[i * 64 + f] = fmaxf(v, 0.0f);
}

// ---------------- global mean pool (same boundary-flush trick) ----------------
__global__ __launch_bounds__(256)
void mean_pool(const float* __restrict__ h, const int* __restrict__ batch,
               const float* __restrict__ cnt, float* __restrict__ out, int N) {
    const int f   = threadIdx.x & 63;
    const int grp = threadIdx.x >> 6;
    const long base = (long)blockIdx.x * 256;
    float a1 = 0.0f;
    int curg = -1;
    for (int j = grp; j < 256; j += 4) {
        long i = base + j;
        if (i >= N) break;
        int g = batch[i];
        if (g != curg) {
            if (curg >= 0)
                atomicAdd(&out[curg * 64 + f], a1 / fmaxf(cnt[curg], 1.0f));
            curg = g; a1 = 0.0f;
        }
        a1 += h[i * 64 + f];
    }
    if (curg >= 0)
        atomicAdd(&out[curg * 64 + f], a1 / fmaxf(cnt[curg], 1.0f));
}

// ---------------------------------------------------------------------------
extern "C" void kernel_launch(void* const* d_in, const int* in_sizes, int n_in,
                              void* d_out, int out_size, void* d_ws, size_t ws_size,
                              hipStream_t stream) {
    const float* x       = (const float*)d_in[0];
    const int*   eidx    = (const int*)d_in[1];
    const int*   batch   = (const int*)d_in[2];
    const float* W1      = (const float*)d_in[3];
    const float* b1      = (const float*)d_in[4];
    const float* alpha1  = (const float*)d_in[5];
    const float* weight1 = (const float*)d_in[6];
    const float* bias1   = (const float*)d_in[7];
    const float* W2      = (const float*)d_in[8];
    const float* b2      = (const float*)d_in[9];
    const float* alpha2  = (const float*)d_in[10];
    const float* weight2 = (const float*)d_in[11];
    const float* bias2   = (const float*)d_in[12];
    float* out = (float*)d_out;

    const int N = in_sizes[2];           // 100000 nodes
    const int E = in_sizes[1] / 2;       // 1.6M edges
    const int IN_DIM = in_sizes[0] / N;  // 128
    const int G = out_size / 64;         // 128 graphs
    const int* src = eidx;
    const int* dst = eidx + E;

    // ---- workspace carve-up (256B aligned) ----
    char* ws = (char*)d_ws;
    size_t off = 0;
    auto carve = [&](size_t bytes) {
        void* p = ws + off;
        off += (bytes + 255) & ~(size_t)255;
        return p;
    };
    float* hA   = (float*)carve((size_t)N * 64 * 4);
    float* hB   = (float*)carve((size_t)N * 64 * 4);
    float* dinv = (float*)carve((size_t)N * 4);
    unsigned short* w1hi = (unsigned short*)carve((size_t)IN_DIM * 64 * 2);
    unsigned short* w1lo = (unsigned short*)carve((size_t)IN_DIM * 64 * 2);
    unsigned short* w2hi = (unsigned short*)carve((size_t)64 * 64 * 2);
    unsigned short* w2lo = (unsigned short*)carve((size_t)64 * 64 * 2);
    float* s1  = (float*)carve((size_t)G * 64 * 4);
    float* s2  = (float*)carve((size_t)G * 64 * 4);
    float* cnt = (float*)carve((size_t)G * 4);
    (void)ws_size; (void)n_in;

    const int B = 256;
    auto blocks = [](long n, int b) { return (unsigned)((n + b - 1) / b); };

    // ---- weight split + fragment swizzle ----
    split_swizzle_weights<<<blocks(IN_DIM * 64, B), B, 0, stream>>>(W1, w1hi, w1lo, IN_DIM);
    split_swizzle_weights<<<blocks(64 * 64, B), B, 0, stream>>>(W2, w2hi, w2lo, 64);

    // ---- degrees -> dinv ----
    fill_f32<<<blocks(N, B), B, 0, stream>>>(dinv, 1.0f, N);  // self-loop
    edge_degree<<<blocks(E, B), B, 0, stream>>>(dst, dinv, E);
    finalize_dinv<<<blocks(N, B), B, 0, stream>>>(dinv, N);

    const unsigned gemmBlocks = blocks(((long)N + 15) / 16, 8);
    const unsigned nodeF4  = blocks((long)N * 16, B);
    const unsigned edgeF4  = blocks((long)E * 16, B);
    const unsigned nodeF   = blocks((long)N * 64, B);
    const unsigned statB   = blocks(N, 256);

    // ================= layer 1 =================
    gemm_wmma_bf16split<128><<<gemmBlocks, 256, 0, stream>>>(x, w1hi, w1lo, hA, N);
    scatter_init<<<nodeF4, B, 0, stream>>>(hA, dinv, b1, hB, N);
    scatter_edges<<<edgeF4, B, 0, stream>>>(hA, src, dst, dinv, hB, E);
    fill_f32<<<blocks(G * 64 * 2 + G, B), B, 0, stream>>>(s1, 0.0f, G * 64 * 2 + G); // s1,s2,cnt contiguous
    graph_stats<<<statB, 256, 0, stream>>>(hB, batch, s1, s2, cnt, N);
    norm_relu<<<nodeF, B, 0, stream>>>(hB, batch, s1, s2, cnt, alpha1, weight1, bias1, hA, N);

    // ================= layer 2 =================
    gemm_wmma_bf16split<64><<<gemmBlocks, 256, 0, stream>>>(hA, w2hi, w2lo, hB, N);
    scatter_init<<<nodeF4, B, 0, stream>>>(hB, dinv, b2, hA, N);
    scatter_edges<<<edgeF4, B, 0, stream>>>(hB, src, dst, dinv, hA, E);
    fill_f32<<<blocks(G * 64 * 2 + G, B), B, 0, stream>>>(s1, 0.0f, G * 64 * 2 + G);
    graph_stats<<<statB, 256, 0, stream>>>(hA, batch, s1, s2, cnt, N);
    norm_relu<<<nodeF, B, 0, stream>>>(hA, batch, s1, s2, cnt, alpha2, weight2, bias2, hB, N);

    // ================= pool =================
    fill_f32<<<blocks(G * 64, B), B, 0, stream>>>(out, 0.0f, G * 64);
    mean_pool<<<statB, 256, 0, stream>>>(hB, batch, cnt, out, N);
}
